// MultiheadStarAttention_53120155516952
// MI455X (gfx1250) — matrix operand order repaired and checked by reading, not compiled, not run on hardware
//
#include <hip/hip_runtime.h>
#include <math.h>

#define L_SEQ 4096
#define BATCH 4
#define DIM   512
#define MROWS (L_SEQ * BATCH)            // 16384
#define SCALE 0.044194173824159216f      // 1/sqrt(512)
#define KCH   32
#define NCHUNK (DIM / KCH)               // 16
#define BROW  40                          // LDS row stride in ushorts (80 B)

typedef __attribute__((ext_vector_type(16))) __bf16        v16bf;
typedef __attribute__((ext_vector_type(8)))  float         v8f;
typedef __attribute__((ext_vector_type(4)))  unsigned int  v4u;
typedef __attribute__((ext_vector_type(4)))  int           v4i;

union FragB16 { v16bf bf; v4u q[2]; };

__device__ inline unsigned short f32_to_bf16_rne(float f) {
  unsigned int u = __float_as_uint(f);
  u += 0x7FFFu + ((u >> 16) & 1u);
  return (unsigned short)(u >> 16);
}
__device__ inline float bf16_to_f32(unsigned short h) {
  return __uint_as_float(((unsigned int)h) << 16);
}

#if __has_builtin(__builtin_amdgcn_global_load_async_to_lds_b128)
#define USE_ASYNC_LDS 1
#endif

// 16-byte global->LDS copy: async on gfx1250 when the builtin exists,
// synchronous fallback otherwise (barriers below are correct for both).
__device__ inline void cp16_async(const unsigned short* g, unsigned short* l) {
#ifdef USE_ASYNC_LDS
  __builtin_amdgcn_global_load_async_to_lds_b128(
      (__attribute__((address_space(1))) v4i*)g,
      (__attribute__((address_space(3))) v4i*)l, 0, 0);
#else
  *(v4u*)l = *(const v4u*)g;
#endif
}
__device__ inline void wait_async_le(int n) {
#ifdef USE_ASYNC_LDS
#if __has_builtin(__builtin_amdgcn_s_wait_asynccnt)
  if (n == 0) __builtin_amdgcn_s_wait_asynccnt(0);
  else        __builtin_amdgcn_s_wait_asynccnt(2);
#else
  if (n == 0) asm volatile("s_wait_asynccnt 0x0" ::: "memory");
  else        asm volatile("s_wait_asynccnt 0x2" ::: "memory");
#endif
#else
  (void)n;  // synchronous fallback: nothing outstanding
#endif
}

// ---------------------------------------------------------------------------
// Elementwise split: f32 -> (bf16 hi, bf16 lo) with lo = bf16(f - f32(hi))
// ---------------------------------------------------------------------------
__global__ __launch_bounds__(256)
void cvt_split(const float* __restrict__ src, unsigned short* __restrict__ hi,
               unsigned short* __restrict__ lo, int n) {
  int i = blockIdx.x * 256 + threadIdx.x;
  if (i < n) {
    float f = src[i];
    unsigned short h = f32_to_bf16_rne(f);
    hi[i] = h;
    lo[i] = f32_to_bf16_rne(f - bf16_to_f32(h));
  }
}

// ---------------------------------------------------------------------------
// C[M,512] = A[M,512] @ W[512,512]^T + bias, bf16x3 error-compensated WMMA.
// Block = 256 threads (8 waves); wave -> 16 rows x 64 cols (4 accumulators);
// block tile = 128 rows x 64 cols. W chunk (64 cols x 32 K, hi+lo) is staged
// in LDS via async global->LDS copies, double-buffered on ASYNCcnt.
// Grid = (M/128, 512/64).
// ---------------------------------------------------------------------------
__global__ __launch_bounds__(256)
void gemm_bf16x3(const unsigned short* __restrict__ Ah,
                 const unsigned short* __restrict__ Al,
                 const unsigned short* __restrict__ Wh,
                 const unsigned short* __restrict__ Wl,
                 const float* __restrict__ bias,
                 float* __restrict__ C) {
  __shared__ __align__(16) unsigned short smem[2][2][64 * BROW];  // 20 KB

  const int tid  = threadIdx.x;
  const int lane = tid & 31;
  const int wave = tid >> 5;
  const int hsel = lane >> 4;
  const int ln16 = lane & 15;
  const int rowBase = blockIdx.x * 128 + wave * 16;
  const int colBase = blockIdx.y * 64;

  // Per-thread copy slot: one 16B segment of the hi tile + one of the lo tile.
  // 64 rows x 32 K ushorts = 256 segments of 8 ushorts; thread t -> seg t.
  const int crow = tid >> 2;   // 0..63
  const int csir = tid & 3;    // 0..3
  const unsigned short* gH = Wh + (size_t)(colBase + crow) * DIM + csir * 8;
  const unsigned short* gL = Wl + (size_t)(colBase + crow) * DIM + csir * 8;
  const int ldst = crow * BROW + csir * 8;

  v8f acc[4];
#pragma unroll
  for (int j = 0; j < 4; ++j) {
    float bj = bias[colBase + j * 16 + ln16];
#pragma unroll
    for (int r = 0; r < 8; ++r) acc[j][r] = bj;
  }

  const unsigned short* aH = Ah + (size_t)(rowBase + ln16) * DIM;
  const unsigned short* aL = Al + (size_t)(rowBase + ln16) * DIM;

  // Prime buffer 0 with chunk 0.
  cp16_async(gH, &smem[0][0][ldst]);
  cp16_async(gL, &smem[0][1][ldst]);

  for (int c = 0; c < NCHUNK; ++c) {
    const int k0  = c * KCH;
    const int buf = c & 1;
    if (c + 1 < NCHUNK) {
      cp16_async(gH + (k0 + KCH), &smem[buf ^ 1][0][ldst]);
      cp16_async(gL + (k0 + KCH), &smem[buf ^ 1][1][ldst]);
      wait_async_le(2);   // current chunk landed; next chunk stays in flight
    } else {
      wait_async_le(0);
    }
    __syncthreads();

    // A fragment: 16-bit A layout = two contiguous runs of 8 K-values per lane
    const int r1 = k0 + hsel * 8;
    const int r2 = k0 + 16 + hsel * 8;
    FragB16 ah, al;
    ah.q[0] = *(const v4u*)(aH + r1);
    ah.q[1] = *(const v4u*)(aH + r2);
    al.q[0] = *(const v4u*)(aL + r1);
    al.q[1] = *(const v4u*)(aL + r2);

#pragma unroll
    for (int j = 0; j < 4; ++j) {
      // B[k,n] = W[n,k]; lane holds two runs of 8 K-values (same as A)
      const int bo = (j * 16 + ln16) * BROW + hsel * 8;
      FragB16 bh, bl;
      bh.q[0] = *(const v4u*)(&smem[buf][0][bo]);
      bh.q[1] = *(const v4u*)(&smem[buf][0][bo + 16]);
      bl.q[0] = *(const v4u*)(&smem[buf][1][bo]);
      bl.q[1] = *(const v4u*)(&smem[buf][1][bo + 16]);
      acc[j] = __builtin_amdgcn_wmma_f32_16x16x32_bf16(false, ah.bf, false, bh.bf,
                                                       (short)0, acc[j], false, false);
      acc[j] = __builtin_amdgcn_wmma_f32_16x16x32_bf16(false, ah.bf, false, bl.bf,
                                                       (short)0, acc[j], false, false);
      acc[j] = __builtin_amdgcn_wmma_f32_16x16x32_bf16(false, al.bf, false, bh.bf,
                                                       (short)0, acc[j], false, false);
    }
    __syncthreads();  // all waves done reading buf before it is refilled
  }

  // D layout: VGPR r, lanes 0-15 -> M=r, lanes 16-31 -> M=r+8; N = lane&15
#pragma unroll
  for (int j = 0; j < 4; ++j) {
    const int col = colBase + j * 16 + ln16;
#pragma unroll
    for (int r = 0; r < 8; ++r)
      C[(size_t)(rowBase + r + 8 * hsel) * DIM + col] = acc[j][r];
  }
}

// ---------------------------------------------------------------------------
// Small exact-fp32 projection: out[rows,512] = A[rows,512] @ W^T + bias
// ---------------------------------------------------------------------------
__global__ __launch_bounds__(256)
void small_proj(const float* __restrict__ A, const float* __restrict__ W,
                const float* __restrict__ bias, float* __restrict__ out, int rows) {
  int idx = blockIdx.x * 256 + threadIdx.x;
  if (idx >= rows * DIM) return;
  int r = idx >> 9, d = idx & (DIM - 1);
  const float* a = A + (size_t)r * DIM;
  const float* w = W + (size_t)d * DIM;
  float acc = bias[d];
  for (int t = 0; t < DIM; ++t) acc += a[t] * w[t];
  out[idx] = acc;
}

// ---------------------------------------------------------------------------
// Ring attention: one wave per (l,b). 5 logits (zero-padded neighbors give
// logit==0, which STILL enters softmax, matching the reference), then a
// 5-way weighted sum.
// ---------------------------------------------------------------------------
__global__ __launch_bounds__(256)
void ring_attn(const float* __restrict__ q, const float* __restrict__ k,
               const float* __restrict__ v, const float* __restrict__ ke,
               const float* __restrict__ ve, const float* __restrict__ kr,
               const float* __restrict__ vr, float* __restrict__ ring) {
  const int lane = threadIdx.x & 31;
  const int gw = blockIdx.x * 8 + (threadIdx.x >> 5);   // gw = l*B + b
  const int b = gw & (BATCH - 1);
  const int l = gw >> 2;

  const float* qv = q + (size_t)gw * DIM;
  const float* keys[5];
  const float* vals[5];
  keys[0] = (l > 0) ? k + (size_t)(gw - BATCH) * DIM : nullptr;
  vals[0] = (l > 0) ? v + (size_t)(gw - BATCH) * DIM : nullptr;
  keys[1] = k + (size_t)gw * DIM;
  vals[1] = v + (size_t)gw * DIM;
  keys[2] = (l < L_SEQ - 1) ? k + (size_t)(gw + BATCH) * DIM : nullptr;
  vals[2] = (l < L_SEQ - 1) ? v + (size_t)(gw + BATCH) * DIM : nullptr;
  keys[3] = ke + (size_t)gw * DIM;
  vals[3] = ve + (size_t)gw * DIM;
  keys[4] = kr + (size_t)b * DIM;
  vals[4] = vr + (size_t)b * DIM;

  float qreg[16];
#pragma unroll
  for (int t = 0; t < 16; ++t) qreg[t] = qv[lane + (t << 5)];

  float logit[5];
#pragma unroll
  for (int w = 0; w < 5; ++w) {
    float p = 0.f;
    if (keys[w]) {
#pragma unroll
      for (int t = 0; t < 16; ++t) p += qreg[t] * keys[w][lane + (t << 5)];
    }
#pragma unroll
    for (int off = 16; off > 0; off >>= 1) p += __shfl_xor(p, off, 32);
    logit[w] = p * SCALE;
  }

  float mx = logit[0];
#pragma unroll
  for (int w = 1; w < 5; ++w) mx = fmaxf(mx, logit[w]);
  float ew[5], den = 0.f;
#pragma unroll
  for (int w = 0; w < 5; ++w) { ew[w] = __expf(logit[w] - mx); den += ew[w]; }
  const float inv = 1.f / den;

#pragma unroll
  for (int t = 0; t < 16; ++t) {
    const int d = lane + (t << 5);
    float acc = 0.f;
#pragma unroll
    for (int w = 0; w < 5; ++w)
      if (vals[w]) acc += ew[w] * vals[w][d];
    ring[(size_t)gw * DIM + d] = acc * inv;
  }
}

// ---------------------------------------------------------------------------
// Star attention: logits, softmax, weighted accumulation
// ---------------------------------------------------------------------------
__global__ __launch_bounds__(256)
void star_logits(const float* __restrict__ relay, const float* __restrict__ k,
                 const float* __restrict__ kr, float* __restrict__ slog) {
  const int b = blockIdx.y;
  const int lane = threadIdx.x & 31;
  const int l = blockIdx.x * 8 + (threadIdx.x >> 5);
  if (l > L_SEQ) return;
  const float* rv = relay + (size_t)b * DIM;
  const float* kv = (l < L_SEQ) ? k + ((size_t)l * BATCH + b) * DIM
                                : kr + (size_t)b * DIM;
  float p = 0.f;
#pragma unroll
  for (int t = 0; t < 16; ++t) {
    const int d = lane + (t << 5);
    p += rv[d] * kv[d];
  }
#pragma unroll
  for (int off = 16; off > 0; off >>= 1) p += __shfl_xor(p, off, 32);
  if (lane == 0) slog[(size_t)b * (L_SEQ + 1) + l] = p * SCALE;
}

__global__ __launch_bounds__(256)
void star_softmax(float* __restrict__ slog) {
  const int b = blockIdx.x;
  float* sl = slog + (size_t)b * (L_SEQ + 1);
  __shared__ float red[8];
  const int tid = threadIdx.x, lane = tid & 31, wid = tid >> 5;

  float mx = -3.4e38f;
  for (int i = tid; i <= L_SEQ; i += 256) mx = fmaxf(mx, sl[i]);
#pragma unroll
  for (int off = 16; off > 0; off >>= 1) mx = fmaxf(mx, __shfl_xor(mx, off, 32));
  if (lane == 0) red[wid] = mx;
  __syncthreads();
  mx = red[0];
#pragma unroll
  for (int i = 1; i < 8; ++i) mx = fmaxf(mx, red[i]);
  __syncthreads();

  float s = 0.f;
  for (int i = tid; i <= L_SEQ; i += 256) {
    float ev = __expf(sl[i] - mx);
    sl[i] = ev;
    s += ev;
  }
#pragma unroll
  for (int off = 16; off > 0; off >>= 1) s += __shfl_xor(s, off, 32);
  if (lane == 0) red[wid] = s;
  __syncthreads();
  s = 0.f;
#pragma unroll
  for (int i = 0; i < 8; ++i) s += red[i];
  const float inv = 1.f / s;
  for (int i = tid; i <= L_SEQ; i += 256) sl[i] *= inv;
}

__global__ __launch_bounds__(512)
void star_accum(const float* __restrict__ slog, const float* __restrict__ v,
                const float* __restrict__ vr, float* __restrict__ star) {
  const int b = blockIdx.x, d = threadIdx.x;
  const float* al = slog + (size_t)b * (L_SEQ + 1);
  float acc = 0.f;
  for (int l = 0; l < L_SEQ; ++l)
    acc += al[l] * v[((size_t)l * BATCH + b) * DIM + d];
  acc += al[L_SEQ] * vr[(size_t)b * DIM + d];
  star[(size_t)b * DIM + d] = acc;
}

// ---------------------------------------------------------------------------
extern "C" void kernel_launch(void* const* d_in, const int* in_sizes, int n_in,
                              void* d_out, int out_size, void* d_ws, size_t ws_size,
                              hipStream_t stream) {
  (void)in_sizes; (void)n_in; (void)out_size; (void)ws_size;
  const float* x     = (const float*)d_in[0];
  const float* e     = (const float*)d_in[1];
  const float* relay = (const float*)d_in[2];
  const float* Wq    = (const float*)d_in[3];
  const float* bq    = (const float*)d_in[4];
  const float* Wk    = (const float*)d_in[5];
  const float* bk    = (const float*)d_in[6];
  const float* Wv    = (const float*)d_in[7];
  const float* bv    = (const float*)d_in[8];
  const float* Wo    = (const float*)d_in[9];
  const float* bo    = (const float*)d_in[10];

  float* out_ring = (float*)d_out;
  float* out_star = out_ring + (size_t)MROWS * DIM;

  const size_t MD  = (size_t)MROWS * DIM;   // 8,388,608
  const size_t WSZ = (size_t)DIM * DIM;     // 262,144

  float* q    = (float*)d_ws;
  float* kbuf = q + MD;
  float* vbuf = kbuf + MD;
  float* ke   = vbuf + MD;
  float* ve   = ke + MD;
  float* ring = ve + MD;
  float* kr   = ring + MD;
  float* vr   = kr + BATCH * DIM;
  float* star = vr + BATCH * DIM;
  float* slog = star + BATCH * DIM;         // BATCH*(L+1) floats

  unsigned short* xh  = (unsigned short*)(slog + (size_t)BATCH * (L_SEQ + 1));
  unsigned short* xl  = xh + MD;
  unsigned short* eh  = xl + MD;
  unsigned short* el  = eh + MD;
  unsigned short* Wqh = el + MD;
  unsigned short* Wql = Wqh + WSZ;
  unsigned short* Wkh = Wql + WSZ;
  unsigned short* Wkl = Wkh + WSZ;
  unsigned short* Wvh = Wkl + WSZ;
  unsigned short* Wvl = Wvh + WSZ;
  unsigned short* Woh = Wvl + WSZ;
  unsigned short* Wol = Woh + WSZ;
  // x splits are dead after the 5 projection GEMMs -> reuse for ring splits
  unsigned short* rh = xh;
  unsigned short* rl = xl;

  const int nMD = (int)MD;
  const int cvtBlk = (nMD + 255) / 256;       // 32768
  const int cvtW   = ((int)WSZ + 255) / 256;  // 1024

  cvt_split<<<cvtBlk, 256, 0, stream>>>(x, xh, xl, nMD);
  cvt_split<<<cvtBlk, 256, 0, stream>>>(e, eh, el, nMD);
  cvt_split<<<cvtW, 256, 0, stream>>>(Wq, Wqh, Wql, (int)WSZ);
  cvt_split<<<cvtW, 256, 0, stream>>>(Wk, Wkh, Wkl, (int)WSZ);
  cvt_split<<<cvtW, 256, 0, stream>>>(Wv, Wvh, Wvl, (int)WSZ);
  cvt_split<<<cvtW, 256, 0, stream>>>(Wo, Woh, Wol, (int)WSZ);

  dim3 ggrid(MROWS / 128, DIM / 64);          // (128, 8)
  gemm_bf16x3<<<ggrid, 256, 0, stream>>>(xh, xl, Wqh, Wql, bq, q);
  gemm_bf16x3<<<ggrid, 256, 0, stream>>>(xh, xl, Wkh, Wkl, bk, kbuf);
  gemm_bf16x3<<<ggrid, 256, 0, stream>>>(xh, xl, Wvh, Wvl, bv, vbuf);
  gemm_bf16x3<<<ggrid, 256, 0, stream>>>(eh, el, Wkh, Wkl, bk, ke);
  gemm_bf16x3<<<ggrid, 256, 0, stream>>>(eh, el, Wvh, Wvl, bv, ve);

  small_proj<<<(BATCH * DIM) / 256, 256, 0, stream>>>(relay, Wk, bk, kr, BATCH);
  small_proj<<<(BATCH * DIM) / 256, 256, 0, stream>>>(relay, Wv, bv, vr, BATCH);

  ring_attn<<<MROWS / 8, 256, 0, stream>>>(q, kbuf, vbuf, ke, ve, kr, vr, ring);

  star_logits<<<dim3(513, BATCH), 256, 0, stream>>>(relay, kbuf, kr, slog);
  star_softmax<<<BATCH, 256, 0, stream>>>(slog);
  star_accum<<<BATCH, 512, 0, stream>>>(slog, vbuf, vr, star);

  cvt_split<<<cvtBlk, 256, 0, stream>>>(ring, rh, rl, nMD);
  gemm_bf16x3<<<ggrid, 256, 0, stream>>>(rh, rl, Woh, Wol, bo, out_ring);
  small_proj<<<(BATCH * DIM) / 256, 256, 0, stream>>>(star, Wo, bo, out_star, BATCH);
}
